// GraphGATPoolNN_83854941487718
// MI455X (gfx1250) — compile-verified, matrix operand order checked by reading, not run on hardware
//
#include <hip/hip_runtime.h>
#include <hip/hip_bf16.h>

// ---------------------------------------------------------------------------
// GraphGATPoolNN for MI455X (gfx1250).
// bf16 WMMA GEMMs (f32 accumulate), 16x64 strip per wave (4 WMMAs per A-frag),
// double-buffered TDM staging of B panels into LDS (s_wait_tensorcnt 1/0);
// f32 hardware atomics for segment softmax / scatter aggregation;
// deterministic handshake matching for EdgePooling.
// ---------------------------------------------------------------------------

typedef __attribute__((ext_vector_type(16))) __bf16 v16bf;
typedef __attribute__((ext_vector_type(8)))  float  v8f;
typedef __attribute__((ext_vector_type(4)))  unsigned int v4u;
typedef __attribute__((ext_vector_type(8)))  int v8i;
typedef __attribute__((ext_vector_type(4)))  int v4i;

#define DEV static __device__ __forceinline__

union BFrag { unsigned short u[16]; v16bf v; };

// float -> bf16 bits, round-to-nearest-even
DEV unsigned short f2bf(float f) {
  unsigned u = __float_as_uint(f);
  u += 0x7FFFu + ((u >> 16) & 1u);
  return (unsigned short)(u >> 16);
}
// order-preserving float<->uint for atomicMax-based segment max
DEV unsigned fenc(float x) {
  unsigned u = __float_as_uint(x);
  return (u & 0x80000000u) ? ~u : (u | 0x80000000u);
}
DEV float fdec(unsigned e) {
  unsigned u = (e & 0x80000000u) ? (e & 0x7FFFFFFFu) : ~e;
  return __uint_as_float(u);
}

// --- Tensor Data Mover: 2D tile of 2-byte elements, global -> LDS -----------
DEV void tdm_load_2d_bf16(unsigned lds_off, const void* gptr,
                          unsigned tensor_w, unsigned tensor_h,
                          unsigned tile_w, unsigned tile_h,
                          unsigned long long row_stride_elems) {
  unsigned long long ga = (unsigned long long)gptr;
  v4u g0;
  g0[0] = 1u;                                           // count=1 valid descriptor
  g0[1] = lds_off;                                      // lds_addr (bytes)
  g0[2] = (unsigned)(ga & 0xFFFFFFFFu);                 // global_addr[31:0]
  g0[3] = (unsigned)((ga >> 32) & 0x01FFFFFFu) | (2u << 30); // addr[56:32], type=2
  v8i g1;
  g1[0] = (int)(1u << 16);                              // data_size=1 (2B), wg_mask=0
  g1[1] = (int)((tensor_w & 0xFFFFu) << 16);            // tensor_dim0[15:0]
  g1[2] = (int)((tensor_w >> 16) | ((tensor_h & 0xFFFFu) << 16));
  g1[3] = (int)((tensor_h >> 16) | ((tile_w & 0xFFFFu) << 16));
  g1[4] = (int)(tile_h & 0xFFFFu);                      // tile_dim1 (tile_dim2=0)
  g1[5] = (int)(row_stride_elems & 0xFFFFFFFFull);      // tensor_dim0_stride lo
  g1[6] = (int)((row_stride_elems >> 32) & 0xFFFFull);  // stride hi (dim1_stride=0)
  g1[7] = 0;
  v4i gz4 = {0, 0, 0, 0};
  v8i gz8 = {0, 0, 0, 0, 0, 0, 0, 0};
  __builtin_amdgcn_tensor_load_to_lds(g0, g1, gz4, gz4, gz8, 0);
}

// ---------------------------------------------------------------------------
// bf16 WMMA GEMM: C[M,N] = relu?(A[M,K] * B[K,N] + bias)
// A: bf16 bits row-major lda=K. B: bf16 bits row-major ldb=N. K % 32 == 0.
// Block: 4 waves of 32. Block tile 64(M) x 64(N); wave w owns rows
// [64*by+16w, +16) x 64 N-columns = 4 accumulators -> 4 WMMAs per A-frag.
// B panels (32x64) double-buffered in LDS via TDM issued by wave 0:
// issue buf[i+1], s_wait_tensorcnt(1) for buf[i] (TDM per-wave in-order).
// ---------------------------------------------------------------------------
__global__ void k_gemm_wmma(const unsigned short* __restrict__ A, int lda,
                            const unsigned short* __restrict__ B, int ldb,
                            const float* __restrict__ bias,
                            float* __restrict__ C, int ldc,
                            const int* __restrict__ pM, int Ncols, int K,
                            int relu) {
  __shared__ unsigned short sB[2][32 * 64];
  const int M    = *pM;
  const int lane = threadIdx.x;
  const int wave = threadIdx.y;
  const int half = lane >> 4;         // 0: lanes 0-15, 1: lanes 16-31
  const int l15  = lane & 15;
  const int tileN0 = blockIdx.x * 64;
  const int tileM  = blockIdx.y * 64 + wave * 16;

  int tw = Ncols - tileN0;
  if (tw > 64) tw = 64;
  if (tw < 0) tw = 0;

  v8f acc0 = {}, acc1 = {}, acc2 = {}, acc3 = {};
  const int nsteps = K >> 5;

  if (wave == 0) {
    tdm_load_2d_bf16((unsigned)(unsigned long long)(&sB[0][0]),
                     B + tileN0, (unsigned)Ncols, (unsigned)K,
                     (unsigned)tw, 32u, (unsigned long long)ldb);
  }

  for (int i = 0; i < nsteps; ++i) {
    const int k0 = i << 5;
    if (wave == 0) {
      if (i + 1 < nsteps) {
        tdm_load_2d_bf16((unsigned)(unsigned long long)(&sB[(i + 1) & 1][0]),
                         B + (size_t)(k0 + 32) * ldb + tileN0,
                         (unsigned)Ncols, (unsigned)K, (unsigned)tw, 32u,
                         (unsigned long long)ldb);
        __builtin_amdgcn_s_wait_tensorcnt(1);  // older load (buf i) complete
      } else {
        __builtin_amdgcn_s_wait_tensorcnt(0);
      }
    }
    __syncthreads();
    const unsigned short* sBc = &sB[i & 1][0];

    // A fragment (16-bit A layout: lanes 0-15 K-base 0, lanes 16-31 K-base 8)
    int r = tileM + l15;
    if (r >= M) r = M - 1;
    if (r < 0) r = 0;
    const unsigned short* Ar = A + (size_t)r * lda + k0 + (half ? 8 : 0);
    BFrag af;
#pragma unroll
    for (int j = 0; j < 8; ++j) {
      const int kk = (j < 4) ? (2 * j) : (16 + 2 * (j - 4));
      af.u[2 * j]     = Ar[kk];
      af.u[2 * j + 1] = Ar[kk + 1];
    }

    // 4 B sub-panels from LDS; B layout mirrors A (lane = N col, VGPR = K)
#pragma unroll
    for (int t = 0; t < 4; ++t) {
      BFrag bf;
#pragma unroll
      for (int j = 0; j < 8; ++j) {
        const int kk = (j < 4) ? (2 * j) : (16 + 2 * (j - 4));
        const int kb = (half ? 8 : 0) + kk;
        bf.u[2 * j]     = sBc[kb * 64 + t * 16 + l15];
        bf.u[2 * j + 1] = sBc[(kb + 1) * 64 + t * 16 + l15];
      }
      if (t == 0)
        acc0 = __builtin_amdgcn_wmma_f32_16x16x32_bf16(false, af.v, false,
                                                       bf.v, (short)0, acc0,
                                                       false, false);
      else if (t == 1)
        acc1 = __builtin_amdgcn_wmma_f32_16x16x32_bf16(false, af.v, false,
                                                       bf.v, (short)0, acc1,
                                                       false, false);
      else if (t == 2)
        acc2 = __builtin_amdgcn_wmma_f32_16x16x32_bf16(false, af.v, false,
                                                       bf.v, (short)0, acc2,
                                                       false, false);
      else
        acc3 = __builtin_amdgcn_wmma_f32_16x16x32_bf16(false, af.v, false,
                                                       bf.v, (short)0, acc3,
                                                       false, false);
    }
    __syncthreads();
  }

  // store: C/D layout -> VGPR r: lanes 0-15 M=r, lanes 16-31 M=8+r
#pragma unroll
  for (int t = 0; t < 4; ++t) {
    const v8f acc = (t == 0) ? acc0 : (t == 1) ? acc1 : (t == 2) ? acc2 : acc3;
    const int col = tileN0 + t * 16 + l15;
    if (col < Ncols) {
#pragma unroll
      for (int r8 = 0; r8 < 8; ++r8) {
        const int row = tileM + (half ? 8 : 0) + r8;
        if (row < M) {
          float v = acc[r8];
          if (bias) v += bias[col];
          if (relu) v = v > 0.f ? v : 0.f;
          C[(size_t)row * ldc + col] = v;
        }
      }
    }
  }
}

// ---------------------------- utility kernels ------------------------------
__global__ void k_cvt_bf16(const float* a, unsigned short* b, int n) {
  int i = blockIdx.x * blockDim.x + threadIdx.x;
  if (i < n) b[i] = f2bf(a[i]);
}
__global__ void k_fill_f32(float* p, float v, int n) {
  int i = blockIdx.x * blockDim.x + threadIdx.x;
  if (i < n) p[i] = v;
}
__global__ void k_fill_u32(unsigned* p, unsigned v, int n) {
  int i = blockIdx.x * blockDim.x + threadIdx.x;
  if (i < n) p[i] = v;
}
__global__ void k_fill_u64(unsigned long long* p, unsigned long long v, int n) {
  int i = blockIdx.x * blockDim.x + threadIdx.x;
  if (i < n) p[i] = v;
}
__global__ void k_fill_i32(int* p, int v, int n) {
  int i = blockIdx.x * blockDim.x + threadIdx.x;
  if (i < n) p[i] = v;
}
__global__ void k_set_i32(int* p, int v) { *p = v; }

__global__ void k_copy_cols(const float* src, int sld, float* dst, int dld,
                            int ncol, int coloff, const int* pM) {
  long long t = (long long)blockIdx.x * blockDim.x + threadIdx.x;
  const int M = *pM;
  if (t >= (long long)M * ncol) return;
  int r = (int)(t / ncol), c = (int)(t % ncol);
  dst[(size_t)r * dld + coloff + c] = src[(size_t)r * sld + c];
}
// dst[:, coloff:coloff+128] = relu(acc + bias)
__global__ void k_bias_relu_cols(const float* acc, const float* bias,
                                 float* dst, int dld, int coloff,
                                 const int* pM) {
  long long t = (long long)blockIdx.x * blockDim.x + threadIdx.x;
  const int M = *pM;
  if (t >= (long long)M * 128) return;
  int r = (int)(t >> 7), c = (int)(t & 127);
  float v = acc[(size_t)r * 128 + c] + bias[c];
  dst[(size_t)r * dld + coloff + c] = v > 0.f ? v : 0.f;
}

// ---------------------------- GAT edge kernels -----------------------------
__global__ void k_node_logits(const float* h, const float* avs,
                              const float* avd, float* als, float* ald,
                              const int* pN) {
  int i = blockIdx.x * blockDim.x + threadIdx.x;
  const int n = *pN;
  int node = i >> 1, hd = i & 1;
  if (node >= n) return;
  const float* hp = h + (size_t)node * 128 + hd * 64;
  const float* av = avs + hd * 64;
  const float* bv = avd + hd * 64;
  float s = 0.f, d = 0.f;
#pragma unroll 8
  for (int c = 0; c < 64; ++c) { float v = hp[c]; s += v * av[c]; d += v * bv[c]; }
  als[node * 2 + hd] = s;
  ald[node * 2 + hd] = d;
}

__global__ void k_edge_alpha(const int* ei, int E_, const int* pN,
                             const float* als, const float* ald, float* alpha,
                             unsigned* menc) {
  int e = blockIdx.x * blockDim.x + threadIdx.x;
  const int n = *pN;
  if (e >= E_ + n) return;
  int s, d;
  if (e < E_) { s = ei[2 * e]; d = ei[2 * e + 1]; } else { s = d = e - E_; }
  if ((unsigned)s >= (unsigned)n || (unsigned)d >= (unsigned)n) {
    alpha[2 * e] = alpha[2 * e + 1] = -1e30f;
    return;
  }
#pragma unroll
  for (int hd = 0; hd < 2; ++hd) {
    float a = als[s * 2 + hd] + ald[d * 2 + hd];
    a = a > 0.f ? a : 0.2f * a;                 // leaky_relu 0.2
    alpha[2 * e + hd] = a;
    atomicMax(&menc[d * 2 + hd], fenc(a));
  }
}

__global__ void k_edge_exp(const int* ei, int E_, const int* pN,
                           const float* alpha, const unsigned* menc,
                           float* eexp, float* ssum) {
  int e = blockIdx.x * blockDim.x + threadIdx.x;
  const int n = *pN;
  if (e >= E_ + n) return;
  int s, d;
  if (e < E_) { s = ei[2 * e]; d = ei[2 * e + 1]; } else { s = d = e - E_; }
  if ((unsigned)s >= (unsigned)n || (unsigned)d >= (unsigned)n) {
    eexp[2 * e] = eexp[2 * e + 1] = 0.f;
    return;
  }
#pragma unroll
  for (int hd = 0; hd < 2; ++hd) {
    float m = fdec(menc[d * 2 + hd]);
    if (!(m > -1e30f && m < 1e30f)) m = 0.f;    // mimic isfinite guard
    float v = expf(alpha[2 * e + hd] - m);
    eexp[2 * e + hd] = v;
    atomicAdd(&ssum[d * 2 + hd], v);
  }
}

__global__ void k_edge_agg(const int* ei, int E_, const int* pN,
                           const float* eexp, const float* ssum,
                           const float* hf, float* out) {
  long long t = (long long)blockIdx.x * blockDim.x + threadIdx.x;
  const int n = *pN;
  if (t >= (long long)(E_ + n) * 128) return;
  int e = (int)(t >> 7), c = (int)(t & 127), hd = c >> 6;
  int s, d;
  if (e < E_) { s = ei[2 * e]; d = ei[2 * e + 1]; } else { s = d = e - E_; }
  if ((unsigned)s >= (unsigned)n || (unsigned)d >= (unsigned)n) return;
  if (c == 0) __builtin_prefetch(&hf[(size_t)s * 128], 0, 1);
  float a = eexp[2 * e + hd] / (ssum[d * 2 + hd] + 1e-16f);
  atomicAdd(&out[(size_t)d * 128 + c], a * hf[(size_t)s * 128 + c]);
}

// ---------------------------- pooling kernels ------------------------------
__global__ void k_edge_score(const float* x, const int* ei, const float* Wp,
                             const float* bp, float* raw, int E_,
                             const int* pN) {
  int e = blockIdx.x * blockDim.x + threadIdx.x;
  if (e >= E_) return;
  const int n = *pN;
  int s = ei[2 * e], d = ei[2 * e + 1];
  if ((unsigned)s >= (unsigned)n || (unsigned)d >= (unsigned)n) {
    raw[e] = -1e30f;
    return;
  }
  const float* xs = x + (size_t)s * 160;
  const float* xd = x + (size_t)d * 160;
  float acc = *bp;
#pragma unroll 8
  for (int c = 0; c < 160; ++c) acc += xs[c] * Wp[c] + xd[c] * Wp[160 + c];
  raw[e] = acc;
}

__global__ void k_smax(const int* ei, const float* raw, unsigned* m, int E_) {
  int e = blockIdx.x * blockDim.x + threadIdx.x;
  if (e >= E_) return;
  atomicMax(&m[ei[2 * e + 1]], fenc(raw[e]));
}
__global__ void k_sexp(const int* ei, const float* raw, const unsigned* m,
                       float* ex, float* ss, int E_) {
  int e = blockIdx.x * blockDim.x + threadIdx.x;
  if (e >= E_) return;
  int d = ei[2 * e + 1];
  float mm = fdec(m[d]);
  if (!(mm > -1e30f && mm < 1e30f)) mm = 0.f;
  float v = expf(raw[e] - mm);
  ex[e] = v;
  atomicAdd(&ss[d], v);
}
__global__ void k_sfin(const int* ei, const float* ex, const float* ss,
                       float* sc, int E_) {
  int e = blockIdx.x * blockDim.x + threadIdx.x;
  if (e >= E_) return;
  sc[e] = ex[e] / (ss[ei[2 * e + 1]] + 1e-16f) + 0.5f;
}

// Handshake matching: propose best incident edge (score-major, edge-minor)
__global__ void k_propose(const int* ei, const float* sc, const int* alive,
                          unsigned long long* best, int E_) {
  int e = blockIdx.x * blockDim.x + threadIdx.x;
  if (e >= E_) return;
  int s = ei[2 * e], d = ei[2 * e + 1];
  if (s == d || !alive[s] || !alive[d]) return;
  unsigned long long pk =
      ((unsigned long long)fenc(sc[e]) << 32) | (unsigned)(~(unsigned)e);
  atomicMax(&best[s], pk);
  atomicMax(&best[d], pk);
}
__global__ void k_accept(const int* ei, const float* sc, int* alive,
                         const unsigned long long* best, int* mate, int* medge,
                         int E_) {
  int e = blockIdx.x * blockDim.x + threadIdx.x;
  if (e >= E_) return;
  int s = ei[2 * e], d = ei[2 * e + 1];
  if (s == d || !alive[s] || !alive[d]) return;
  unsigned long long pk =
      ((unsigned long long)fenc(sc[e]) << 32) | (unsigned)(~(unsigned)e);
  if (best[s] == pk && best[d] == pk) {   // unique winner per node
    mate[s] = d; mate[d] = s;
    medge[s] = e; medge[d] = e;
    alive[s] = 0; alive[d] = 0;
  }
}
// Deterministic sequential cluster-id compaction (single thread, n ~ 40k).
__global__ void k_assign_ids(const int* pN, const int* mate, const int* medge,
                             const float* sc, int* cluster, float* cscore,
                             int* cntOut) {
  const int n = *pN;
  int id = 0;
  for (int i = 0; i < n; ++i) {
    int m = mate[i];
    if (m >= 0) {
      if (m > i) {                       // first endpoint owns the pair
        cluster[i] = id; cluster[m] = id;
        cscore[id] = sc[medge[i]];
        ++id;
      }
    } else {
      cluster[i] = id; cscore[id] = 1.0f; ++id;
    }
  }
  *cntOut = id;
}
__global__ void k_remap(const int* ei, const int* cluster, int* eo, int E_,
                        const int* pN) {
  int e = blockIdx.x * blockDim.x + threadIdx.x;
  if (e >= E_) return;
  const int n = *pN;
  int s = ei[2 * e], d = ei[2 * e + 1];
  eo[2 * e]     = ((unsigned)s < (unsigned)n) ? cluster[s] : 0;
  eo[2 * e + 1] = ((unsigned)d < (unsigned)n) ? cluster[d] : 0;
}
__global__ void k_pool_agg(const float* x, const int* cluster, float* nx,
                           const int* pN) {
  long long t = (long long)blockIdx.x * blockDim.x + threadIdx.x;
  const int n = *pN;
  if (t >= (long long)n * 160) return;
  int i = (int)(t / 160), c = (int)(t % 160);
  atomicAdd(&nx[(size_t)cluster[i] * 160 + c], x[(size_t)i * 160 + c]);
}
__global__ void k_pool_scale(float* nx, const float* cs, const int* pNn) {
  long long t = (long long)blockIdx.x * blockDim.x + threadIdx.x;
  const int nn = *pNn;
  if (t >= (long long)nn * 160) return;
  nx[t] *= cs[t / 160];
}
__global__ void k_unpool(const float* xc, const int* cluster, const float* cs,
                         float* xo, const int* pN) {
  long long t = (long long)blockIdx.x * blockDim.x + threadIdx.x;
  const int n = *pN;
  if (t >= (long long)n * 128) return;
  int i = (int)(t >> 7), c = (int)(t & 127);
  int j = cluster[i];
  xo[t] = xc[(size_t)j * 128 + c] / cs[j];
}
__global__ void k_sigmoid(const float* a, float* o, int n) {
  int i = blockIdx.x * blockDim.x + threadIdx.x;
  if (i < n) o[i] = 1.f / (1.f + expf(-a[i]));
}

// ---------------------------------------------------------------------------
extern "C" void kernel_launch(void* const* d_in, const int* in_sizes, int n_in,
                              void* d_out, int out_size, void* d_ws,
                              size_t ws_size, hipStream_t stream) {
  const int N = 40000, F = 32, E = 640000, HID = 128, D1 = 160, D2 = 288;
  const int ET = E + N;

  const float* x   = (const float*)d_in[0];
  const int*   ei0 = (const int*)d_in[1];
  const float *W1 = (const float*)d_in[2],  *a1s = (const float*)d_in[3],
              *a1d = (const float*)d_in[4], *b1 = (const float*)d_in[5];
  const float *W2 = (const float*)d_in[6],  *a2s = (const float*)d_in[7],
              *a2d = (const float*)d_in[8], *b2 = (const float*)d_in[9];
  const float *W3 = (const float*)d_in[10], *a3s = (const float*)d_in[11],
              *a3d = (const float*)d_in[12], *b3 = (const float*)d_in[13];
  const float *W4 = (const float*)d_in[14], *a4s = (const float*)d_in[15],
              *a4d = (const float*)d_in[16], *b4 = (const float*)d_in[17];
  const float *Wp[3] = {(const float*)d_in[18], (const float*)d_in[20],
                        (const float*)d_in[22]};
  const float *bp[3] = {(const float*)d_in[19], (const float*)d_in[21],
                        (const float*)d_in[23]};
  const float *Wf1 = (const float*)d_in[24], *bf1 = (const float*)d_in[25];
  const float *Wf2 = (const float*)d_in[26], *bf2 = (const float*)d_in[27];

  // ---- workspace bump allocator ----
  char* wptr = (char*)d_ws;
  auto alloc = [&](size_t b) -> void* {
    void* p = (void*)wptr;
    wptr += (b + 255) & ~(size_t)255;
    return p;
  };
  unsigned short* Ab  = (unsigned short*)alloc((size_t)N * D2 * 2); // GEMM A scratch
  unsigned short* W1b = (unsigned short*)alloc((size_t)F * HID * 2);
  unsigned short* W2b = (unsigned short*)alloc((size_t)D1 * HID * 2);
  unsigned short* W3b = (unsigned short*)alloc((size_t)D2 * HID * 2);
  unsigned short* W4b = (unsigned short*)alloc((size_t)D2 * HID * 2);
  unsigned short* Wf1b = (unsigned short*)alloc((size_t)D1 * 64 * 2);
  unsigned short* Wf2b = (unsigned short*)alloc((size_t)96 * 1 * 2);
  float* hmat = (float*)alloc((size_t)N * HID * 4);   // h = A*W
  float* agg  = (float*)alloc((size_t)N * HID * 4);   // attention aggregation
  float* hu   = (float*)alloc((size_t)N * HID * 4);   // relu(h)+unpool ping
  float* als  = (float*)alloc((size_t)N * 2 * 4);
  float* ald  = (float*)alloc((size_t)N * 2 * 4);
  unsigned* menc = (unsigned*)alloc((size_t)N * 2 * 4);
  float* ssum = (float*)alloc((size_t)N * 2 * 4);
  float* alpha = (float*)alloc((size_t)ET * 2 * 4);
  float* eexp  = (float*)alloc((size_t)ET * 2 * 4);
  float* xlvl[4];
  for (int l = 0; l < 4; ++l) xlvl[l] = (float*)alloc((size_t)N * D1 * 4);
  float* catD2 = (float*)alloc((size_t)N * D2 * 4);
  float* raw = (float*)alloc((size_t)E * 4);
  float* sce = (float*)alloc((size_t)E * 4);
  float* sex = (float*)alloc((size_t)E * 4);
  unsigned* sm = (unsigned*)alloc((size_t)N * 4);
  float* ssc = (float*)alloc((size_t)N * 4);
  unsigned long long* best = (unsigned long long*)alloc((size_t)N * 8);
  int* alive = (int*)alloc((size_t)N * 4);
  int* mate  = (int*)alloc((size_t)N * 4);
  int* medge = (int*)alloc((size_t)N * 4);
  int* clus[3]; float* csc[3];
  for (int l = 0; l < 3; ++l) {
    clus[l] = (int*)alloc((size_t)N * 4);
    csc[l]  = (float*)alloc((size_t)N * 4);
  }
  int* eibuf[3];
  for (int l = 0; l < 3; ++l) eibuf[l] = (int*)alloc((size_t)E * 2 * 4);
  int* cnt = (int*)alloc(64);
  float* f1 = (float*)alloc((size_t)N * 64 * 4);
  float* cat96 = (float*)alloc((size_t)N * 96 * 4);
  float* logits = (float*)alloc((size_t)N * 4);

  const dim3 B256(256);
  auto G = [](long long n) { return dim3((unsigned)((n + 255) / 256)); };

  // counts
  k_set_i32<<<dim3(1), dim3(1), 0, stream>>>(cnt + 0, N);

  // weight conversions (bf16)
  k_cvt_bf16<<<G(F * HID), B256, 0, stream>>>(W1, W1b, F * HID);
  k_cvt_bf16<<<G(D1 * HID), B256, 0, stream>>>(W2, W2b, D1 * HID);
  k_cvt_bf16<<<G(D2 * HID), B256, 0, stream>>>(W3, W3b, D2 * HID);
  k_cvt_bf16<<<G(D2 * HID), B256, 0, stream>>>(W4, W4b, D2 * HID);
  k_cvt_bf16<<<G(D1 * 64), B256, 0, stream>>>(Wf1, Wf1b, D1 * 64);
  k_cvt_bf16<<<G(96), B256, 0, stream>>>(Wf2, Wf2b, 96);

  // GAT layer: agg <- attention-aggregated (no bias/relu yet)
  auto gat = [&](const float* Ain, int K, const unsigned short* Wb,
                 const float* avs, const float* avd, const int* ei,
                 const int* pN) {
    k_cvt_bf16<<<G((long long)N * K), B256, 0, stream>>>(Ain, Ab, N * K);
    dim3 gg((HID + 63) / 64, (N + 63) / 64), gb(32, 4);
    k_gemm_wmma<<<gg, gb, 0, stream>>>(Ab, K, Wb, HID, nullptr, hmat, HID, pN,
                                       HID, K, 0);
    k_node_logits<<<G((long long)N * 2), B256, 0, stream>>>(hmat, avs, avd,
                                                            als, ald, pN);
    k_fill_u32<<<G(N * 2), B256, 0, stream>>>(menc, 0u, N * 2);
    k_fill_f32<<<G(N * 2), B256, 0, stream>>>(ssum, 0.f, N * 2);
    k_edge_alpha<<<G(ET), B256, 0, stream>>>(ei, E, pN, als, ald, alpha, menc);
    k_edge_exp<<<G(ET), B256, 0, stream>>>(ei, E, pN, alpha, menc, eexp, ssum);
    k_fill_f32<<<G((long long)N * HID), B256, 0, stream>>>(agg, 0.f, N * HID);
    k_edge_agg<<<G((long long)ET * HID), B256, 0, stream>>>(ei, E, pN, eexp,
                                                            ssum, hmat, agg);
  };

  // ---- GAT1: x1 = [x | relu(gat1(x)+b1)] ----
  gat(x, F, W1b, a1s, a1d, ei0, cnt + 0);
  k_copy_cols<<<G((long long)N * F), B256, 0, stream>>>(x, F, xlvl[0], D1, F,
                                                        0, cnt + 0);
  k_bias_relu_cols<<<G((long long)N * HID), B256, 0, stream>>>(
      agg, b1, xlvl[0], D1, F, cnt + 0);

  // ---- 3 EdgePooling levels ----
  const int* curei = ei0;
  for (int l = 0; l < 3; ++l) {
    const int* pN = cnt + l;
    int* pNn = cnt + l + 1;
    // edge scores + per-dst softmax + 0.5
    k_edge_score<<<G(E), B256, 0, stream>>>(xlvl[l], curei, Wp[l], bp[l], raw,
                                            E, pN);
    k_fill_u32<<<G(N), B256, 0, stream>>>(sm, 0u, N);
    k_fill_f32<<<G(N), B256, 0, stream>>>(ssc, 0.f, N);
    k_smax<<<G(E), B256, 0, stream>>>(curei, raw, sm, E);
    k_sexp<<<G(E), B256, 0, stream>>>(curei, raw, sm, sex, ssc, E);
    k_sfin<<<G(E), B256, 0, stream>>>(curei, sex, ssc, sce, E);
    // handshake maximal matching (deterministic), 6 rounds
    k_fill_i32<<<G(N), B256, 0, stream>>>(alive, 1, N);
    k_fill_i32<<<G(N), B256, 0, stream>>>(mate, -1, N);
    k_fill_i32<<<G(N), B256, 0, stream>>>(medge, 0, N);
    for (int r = 0; r < 6; ++r) {
      k_fill_u64<<<G(N), B256, 0, stream>>>(best, 0ull, N);
      k_propose<<<G(E), B256, 0, stream>>>(curei, sce, alive, best, E);
      k_accept<<<G(E), B256, 0, stream>>>(curei, sce, alive, best, mate, medge,
                                          E);
    }
    k_assign_ids<<<dim3(1), dim3(1), 0, stream>>>(pN, mate, medge, sce,
                                                  clus[l], csc[l], pNn);
    // new_x = segment_sum(x, cluster) * new_score
    k_fill_f32<<<G((long long)N * D1), B256, 0, stream>>>(xlvl[l + 1], 0.f,
                                                          N * D1);
    k_pool_agg<<<G((long long)N * D1), B256, 0, stream>>>(xlvl[l], clus[l],
                                                          xlvl[l + 1], pN);
    k_pool_scale<<<G((long long)N * D1), B256, 0, stream>>>(xlvl[l + 1],
                                                            csc[l], pNn);
    // remap edges (note: duplicates not coalesced)
    k_remap<<<G(E), B256, 0, stream>>>(curei, clus[l], eibuf[l], E, pN);
    curei = eibuf[l];
  }

  // ---- GAT2/3/4 at the pooled level ----
  const int* pN3 = cnt + 3;
  gat(xlvl[3], D1, W2b, a2s, a2d, curei, pN3);
  k_bias_relu_cols<<<G((long long)N * HID), B256, 0, stream>>>(agg, b2, hu,
                                                               HID, 0, pN3);
  k_copy_cols<<<G((long long)N * D1), B256, 0, stream>>>(xlvl[3], D1, catD2,
                                                         D2, D1, 0, pN3);
  k_copy_cols<<<G((long long)N * HID), B256, 0, stream>>>(hu, HID, catD2, D2,
                                                          HID, D1, pN3);
  gat(catD2, D2, W3b, a3s, a3d, curei, pN3);
  k_bias_relu_cols<<<G((long long)N * HID), B256, 0, stream>>>(agg, b3, hu,
                                                               HID, 0, pN3);
  k_copy_cols<<<G((long long)N * HID), B256, 0, stream>>>(hu, HID, catD2, D2,
                                                          HID, D1, pN3);
  gat(catD2, D2, W4b, a4s, a4d, curei, pN3);
  k_bias_relu_cols<<<G((long long)N * HID), B256, 0, stream>>>(agg, b4, hu,
                                                               HID, 0, pN3);

  // ---- unpool (levels 2,1,0): x = (x / score)[cluster] ----
  float* ua = hu;
  float* ub = agg;   // reuse as ping-pong
  for (int l = 2; l >= 0; --l) {
    k_unpool<<<G((long long)N * HID), B256, 0, stream>>>(ua, clus[l], csc[l],
                                                         ub, cnt + l);
    float* t = ua; ua = ub; ub = t;
  }
  // ua now holds [N,128] at level 0.

  // ---- final MLP ----
  // xlvl[0][:, :32] still equals x; overwrite cols 32..160 with unpooled h.
  k_copy_cols<<<G((long long)N * HID), B256, 0, stream>>>(ua, HID, xlvl[0], D1,
                                                          HID, F, cnt + 0);
  k_cvt_bf16<<<G((long long)N * D1), B256, 0, stream>>>(xlvl[0], Ab, N * D1);
  {
    dim3 gg(1, (N + 63) / 64), gb(32, 4);
    k_gemm_wmma<<<gg, gb, 0, stream>>>(Ab, D1, Wf1b, 64, bf1, f1, 64, cnt + 0,
                                       64, D1, 1);
  }
  k_copy_cols<<<G((long long)N * F), B256, 0, stream>>>(x, F, cat96, 96, F, 0,
                                                        cnt + 0);
  k_copy_cols<<<G((long long)N * 64), B256, 0, stream>>>(f1, 64, cat96, 96, 64,
                                                         F, cnt + 0);
  k_cvt_bf16<<<G((long long)N * 96), B256, 0, stream>>>(cat96, Ab, N * 96);
  {
    dim3 gg(1, (N + 63) / 64), gb(32, 4);
    k_gemm_wmma<<<gg, gb, 0, stream>>>(Ab, 96, Wf2b, 1, bf2, logits, 1,
                                       cnt + 0, 1, 96, 0);
  }
  k_sigmoid<<<G(N), B256, 0, stream>>>(logits, (float*)d_out, N);
  (void)in_sizes; (void)n_in; (void)out_size; (void)ws_size;
}
